// DenseLinearAttention_62302795596508
// MI455X (gfx1250) — matrix operand order repaired and checked by reading, not compiled, not run on hardware
//
#include <hip/hip_runtime.h>
#include <cstdint>
#include <cstddef>

typedef __attribute__((ext_vector_type(16))) __bf16 v16bf;
typedef __attribute__((ext_vector_type(8)))  float  v8f;
typedef __attribute__((ext_vector_type(4)))  unsigned int v4u;
typedef __attribute__((ext_vector_type(8)))  int v8i;
typedef __attribute__((ext_vector_type(4)))  int v4i;

#define B_  8
#define T_  4096
#define D_  1024
#define H_  64
#define C_  64
#define NC_ (T_ / C_)
#define ROWS_ (B_ * T_)

#if __has_builtin(__builtin_amdgcn_tensor_load_to_lds) && \
    __has_builtin(__builtin_amdgcn_s_wait_tensorcnt)
#define HAVE_TDM 1
#else
#define HAVE_TDM 0
#endif

__device__ __forceinline__ v8f zero8() {
  v8f z = {0.f, 0.f, 0.f, 0.f, 0.f, 0.f, 0.f, 0.f};
  return z;
}

__device__ __forceinline__ float phi_act(float x) {
  // elu(x) + 1
  return x > 0.f ? x + 1.f : __expf(x);
}

// A fragment of a 16x32 bf16 tile stored row-major: A[m][k] = p[m*ld + k].
// frag[j] = A[m][hi*16 + j] (j<16)
// -> two contiguous 16B runs per lane (vectorizes to ds_load_b128).
__device__ __forceinline__ v16bf load_a_row(const __bf16* p, int ld, int lane) {
  int m = lane & 15, hi = lane >> 4;
  v16bf f;
#pragma unroll
  for (int g = 0; g < 2; ++g)
#pragma unroll
    for (int j = 0; j < 8; ++j)
      f[g * 8 + j] = p[m * ld + hi * 16 + g * 8 + j];
  return f;
}

// B fragment of a 32x16 bf16 tile stored N-major (transposed):
// B[k][n] = pT[n*ld + k]. frag[j] = pT[n*ld + hi*16 + j]
// -> one contiguous 32B run per lane (2x ds_load_b128).
__device__ __forceinline__ v16bf load_bT(const __bf16* pT, int ld, int lane) {
  int n = lane & 15, hi = lane >> 4;
  v16bf f;
#pragma unroll
  for (int j = 0; j < 16; ++j)
    f[j] = pT[n * ld + hi * 16 + j];
  return f;
}

__device__ __forceinline__ v8f wmma_bf16(v16bf a, v16bf b, v8f c) {
  return __builtin_amdgcn_wmma_f32_16x16x32_bf16(false, a, false, b, (short)0, c,
                                                 false, false);
}

#if HAVE_TDM
// TDM: DMA a contiguous run of n8*8 bytes from global to LDS.
// 1-D descriptor: data_size=8B, tensor_dim0 = tile_dim0 = n8, stride = n8.
__device__ __forceinline__ void tdm_load_1d(const void* gptr, void* lptr, int n8) {
  unsigned long long ga = (unsigned long long)(size_t)gptr;
  unsigned int la = (unsigned int)(size_t)lptr;
  v4u g0;
  g0[0] = 1u;                                        // count=1 (valid), user mode
  g0[1] = la;                                        // lds_addr
  g0[2] = (unsigned int)(ga & 0xffffffffu);          // global_addr[31:0]
  g0[3] = (unsigned int)((ga >> 32) & 0x01ffffffu)   // global_addr[56:32]
          | (2u << 30);                              // type = 2 ("image")
  v8i g1;
  g1[0] = 0x00030000;                                // data_size = 3 (8 bytes)
  g1[1] = (int)((unsigned)(n8 & 0xffff) << 16);      // tensor_dim0[15:0]
  g1[2] = (int)(((unsigned)n8 >> 16) & 0xffffu) | (1 << 16);  // dim0 hi | tensor_dim1=1
  g1[3] = (int)((unsigned)(n8 & 0xffff) << 16);      // tile_dim0
  g1[4] = 0;                                         // tile_dim1/2 unused
  g1[5] = n8;                                        // tensor_dim0_stride[31:0]
  g1[6] = 0;
  g1[7] = 0;
  v4i z4 = {0, 0, 0, 0};
#if __clang_major__ >= 23
  v8i z8 = {0, 0, 0, 0, 0, 0, 0, 0};
  __builtin_amdgcn_tensor_load_to_lds(g0, g1, z4, z4, z8, 0);
#else
  __builtin_amdgcn_tensor_load_to_lds(g0, g1, z4, z4, 0);
#endif
}
#endif

// ---------------------------------------------------------------------------
// K0: convert weights f32 -> bf16 (Wq/Wk/Wv are D x H, Wo is H x D; all 65536)
// ---------------------------------------------------------------------------
__global__ void __launch_bounds__(256) convert_weights_kernel(
    const float* __restrict__ Wq, const float* __restrict__ Wk,
    const float* __restrict__ Wv, const float* __restrict__ Wo,
    __bf16* __restrict__ wqb, __bf16* __restrict__ wkb,
    __bf16* __restrict__ wvb, __bf16* __restrict__ wob) {
  int i = blockIdx.x * blockDim.x + threadIdx.x;
  if (i < D_ * H_) {
    wqb[i] = (__bf16)Wq[i];
    wkb[i] = (__bf16)Wk[i];
    wvb[i] = (__bf16)Wv[i];
    wob[i] = (__bf16)Wo[i];
  }
}

// ---------------------------------------------------------------------------
// K1: fused QKV projection. Each block: 64 rows x 64 cols of Q,K,V.
// ---------------------------------------------------------------------------
__global__ void __launch_bounds__(256) qkv_kernel(
    const float* __restrict__ x,
    const __bf16* __restrict__ wqb, const __bf16* __restrict__ wkb,
    const __bf16* __restrict__ wvb,
    __bf16* __restrict__ qb, __bf16* __restrict__ kb, __bf16* __restrict__ vb) {
  __shared__ alignas(16) __bf16 xs[64 * 32];        // 64 rows x 32 K, row-major
  __shared__ alignas(16) __bf16 wT[3][H_ * 32];     // N-major: wT[n*32 + k]
  const int tid = threadIdx.x, wave = tid >> 5, lane = tid & 31;
  const int row0 = blockIdx.x * 64;
  const int n = lane & 15, hi = lane >> 4;

  v8f acc[2][3];
#pragma unroll
  for (int i = 0; i < 2; ++i)
#pragma unroll
    for (int m = 0; m < 3; ++m) acc[i][m] = zero8();

  const __bf16* wsrc[3] = {wqb, wkb, wvb};
  const int xr = tid >> 2;          // 0..63
  const int xc = (tid & 3) * 8;     // 0,8,16,24

  for (int kk = 0; kk < D_ / 32; ++kk) {
    {
      const float4* xp =
          (const float4*)(x + (size_t)(row0 + xr) * D_ + kk * 32 + xc);
      float4 f0 = xp[0], f1 = xp[1];
      __bf16* d = xs + xr * 32 + xc;
      d[0] = (__bf16)f0.x; d[1] = (__bf16)f0.y;
      d[2] = (__bf16)f0.z; d[3] = (__bf16)f0.w;
      d[4] = (__bf16)f1.x; d[5] = (__bf16)f1.y;
      d[6] = (__bf16)f1.z; d[7] = (__bf16)f1.w;
    }
#pragma unroll
    for (int m = 0; m < 3; ++m) {
      uint4 wv = *(const uint4*)(wsrc[m] + (size_t)kk * 2048 + tid * 8);
      const __bf16* e = (const __bf16*)&wv;
      int k = (tid * 8) >> 6, n0 = (tid * 8) & 63;
#pragma unroll
      for (int j = 0; j < 8; ++j) wT[m][(n0 + j) * 32 + k] = e[j];
    }
    if (kk + 1 < D_ / 32)  // global_prefetch_b8 for next x slab
      __builtin_prefetch(x + (size_t)(row0 + xr) * D_ + (kk + 1) * 32 + xc, 0, 1);
    __syncthreads();
#pragma unroll
    for (int pi = 0; pi < 2; ++pi) {
      int p = wave + pi * 8, mt = p >> 2, nt = p & 3;
      v16bf a = load_a_row(xs + mt * 16 * 32, 32, lane);
#pragma unroll
      for (int m = 0; m < 3; ++m) {
        v16bf b = load_bT(&wT[m][nt * 16 * 32], 32, lane);
        acc[pi][m] = wmma_bf16(a, b, acc[pi][m]);
      }
    }
    __syncthreads();
  }
#pragma unroll
  for (int pi = 0; pi < 2; ++pi) {
    int p = wave + pi * 8, mt = p >> 2, nt = p & 3;
#pragma unroll
    for (int m = 0; m < 3; ++m) {
      __bf16* dst = (m == 0) ? qb : (m == 1) ? kb : vb;
#pragma unroll
      for (int r = 0; r < 8; ++r) {
        float v = acc[pi][m][r];
        if (m < 2) v = phi_act(v);
        int row = row0 + mt * 16 + r + hi * 8;
        int col = nt * 16 + n;
        dst[(size_t)row * H_ + col] = (__bf16)v;
      }
    }
  }
}

// ---------------------------------------------------------------------------
// K2: per-chunk state S_c = K_c^T V_c (64x64 f32) and z_c = colsum(K_c).
// ---------------------------------------------------------------------------
__global__ void __launch_bounds__(256) chunk_state_kernel(
    const __bf16* __restrict__ kb, const __bf16* __restrict__ vb,
    float* __restrict__ Sc, float* __restrict__ zc) {
  __shared__ alignas(16) __bf16 kT[C_ * H_];  // kT[i*64 + s] = K[s][i]
  __shared__ alignas(16) __bf16 vT[C_ * H_];  // vT[j*64 + s] = V[s][j]
  const int tid = threadIdx.x, wave = tid >> 5, lane = tid & 31;
  const int blk = blockIdx.x;
  const size_t base = (size_t)blk * C_ * H_;
  for (int i0 = tid * 8; i0 < C_ * H_; i0 += 256 * 8) {
    uint4 a = *(const uint4*)(kb + base + i0);
    uint4 b = *(const uint4*)(vb + base + i0);
    const __bf16* ea = (const __bf16*)&a;
    const __bf16* eb = (const __bf16*)&b;
    int t = i0 >> 6, h0 = i0 & 63;
#pragma unroll
    for (int j = 0; j < 8; ++j) {
      kT[(h0 + j) * 64 + t] = ea[j];
      vT[(h0 + j) * 64 + t] = eb[j];
    }
  }
  __syncthreads();
  const int n = lane & 15, hi = lane >> 4;
  float* Sout = Sc + (size_t)blk * H_ * H_;
#pragma unroll
  for (int pi = 0; pi < 2; ++pi) {
    int p = wave + pi * 8, mt = p >> 2, nt = p & 3;
    v8f acc = zero8();
#pragma unroll
    for (int ks = 0; ks < 2; ++ks) {
      v16bf a = load_a_row(kT + mt * 16 * 64 + ks * 32, 64, lane);  // A = K^T
      v16bf b = load_bT(vT + nt * 16 * 64 + ks * 32, 64, lane);     // B = V
      acc = wmma_bf16(a, b, acc);
    }
#pragma unroll
    for (int r = 0; r < 8; ++r) {
      int row = mt * 16 + r + hi * 8, col = nt * 16 + n;
      Sout[row * H_ + col] = acc[r];
    }
  }
  if (tid < H_) {
    float s = 0.f;
    for (int t = 0; t < C_; ++t) s += (float)kT[tid * 64 + t];
    zc[(size_t)blk * H_ + tid] = s;
  }
}

// ---------------------------------------------------------------------------
// K3: serial exclusive prefix over chunks (per batch).
// ---------------------------------------------------------------------------
__global__ void __launch_bounds__(256) prefix_kernel(
    const float* __restrict__ Sc, const float* __restrict__ zc,
    __bf16* __restrict__ Sp, float* __restrict__ zp) {
  const int b = blockIdx.x, tid = threadIdx.x;
  float s[16];
#pragma unroll
  for (int e = 0; e < 16; ++e) s[e] = 0.f;
  float zrun = 0.f;
  for (int c = 0; c < NC_; ++c) {
    const size_t blk = (size_t)b * NC_ + c;
    __bf16* outp = Sp + blk * (size_t)(H_ * H_);
#pragma unroll
    for (int e = 0; e < 16; ++e) outp[tid * 16 + e] = (__bf16)s[e];
    if (tid < H_) zp[blk * H_ + tid] = zrun;
    const float* Sin = Sc + blk * (size_t)(H_ * H_);
#pragma unroll
    for (int e = 0; e < 16; ++e) s[e] += Sin[tid * 16 + e];
    if (tid < H_) zrun += zc[blk * H_ + tid];
  }
}

// ---------------------------------------------------------------------------
// K4: per-chunk output. P = mask(Q V^T); denom = rowsum(mask(Q K^T)) + Q.zpref
// O = P K + Q Sp^T; o = O/denom; out_tile = o @ W_o (fused, f32 result).
// Tile staging uses the Tensor Data Mover when available.
// ---------------------------------------------------------------------------
__global__ void __launch_bounds__(256) chunk_out_kernel(
    const __bf16* __restrict__ qb, const __bf16* __restrict__ kb,
    const __bf16* __restrict__ vb, const __bf16* __restrict__ Sp,
    const float* __restrict__ zp, const __bf16* __restrict__ wob,
    float* __restrict__ outp) {
  __shared__ alignas(16) __bf16 qsh[C_ * H_];    // Q natural
  __shared__ alignas(16) __bf16 ksh[C_ * H_];    // K natural (= N-major K^T)
  __shared__ alignas(16) __bf16 vshkT[C_ * H_];  // phase1: V natural; phase2: K^T
  __shared__ alignas(16) __bf16 sps[H_ * H_];    // S_prefix natural
  __shared__ alignas(16) __bf16 psh[C_ * C_];    // P (bf16)
  __shared__ alignas(16) __bf16 osh[C_ * H_];    // normalized o (bf16)
  __shared__ alignas(16) __bf16 wshT[64 * H_];   // W_o tile, N-major
  __shared__ float denom[C_], zsh[H_];
  const int tid = threadIdx.x, wave = tid >> 5, lane = tid & 31;
  const int n = lane & 15, hi = lane >> 4;
  const int blk = blockIdx.x;
  const size_t base = (size_t)blk * C_ * H_;

#if HAVE_TDM
  if (wave == 0) {  // one wave drives 4 TDM tile DMAs (EXEC is ignored by TDM)
    tdm_load_1d(qb + base, qsh, (C_ * H_ * 2) / 8);
    tdm_load_1d(kb + base, ksh, (C_ * H_ * 2) / 8);
    tdm_load_1d(vb + base, vshkT, (C_ * H_ * 2) / 8);
    tdm_load_1d(Sp + (size_t)blk * (H_ * H_), sps, (H_ * H_ * 2) / 8);
    __builtin_amdgcn_s_wait_tensorcnt(0);
  }
#else
  {
    const uint4* gq = (const uint4*)(qb + base);
    const uint4* gk = (const uint4*)(kb + base);
    const uint4* gv = (const uint4*)(vb + base);
    const uint4* gs = (const uint4*)(Sp + (size_t)blk * (H_ * H_));
    uint4* sq = (uint4*)qsh;
    uint4* sk = (uint4*)ksh;
    uint4* sv = (uint4*)vshkT;
    uint4* ss = (uint4*)sps;
    for (int i = tid; i < (C_ * H_) / 8; i += 256) {
      sq[i] = gq[i]; sk[i] = gk[i]; sv[i] = gv[i]; ss[i] = gs[i];
    }
  }
#endif
  if (tid < H_) {
    zsh[tid] = zp[(size_t)blk * H_ + tid];
    denom[tid] = 0.f;
  }
  __syncthreads();

  // Phase 1: P = mask(Q V^T); denom += rowsum(mask(Q K^T))
#pragma unroll
  for (int pi = 0; pi < 2; ++pi) {
    int p = wave + pi * 8, mt = p >> 2, nt = p & 3;
    v8f aP = zero8(), aD = zero8();
#pragma unroll
    for (int ks = 0; ks < 2; ++ks) {
      v16bf aq = load_a_row(qsh + mt * 16 * H_ + ks * 32, H_, lane);
      v16bf bv = load_bT(vshkT + nt * 16 * H_ + ks * 32, H_, lane);  // B = V^T
      aP = wmma_bf16(aq, bv, aP);
      v16bf bk = load_bT(ksh + nt * 16 * H_ + ks * 32, H_, lane);    // B = K^T
      aD = wmma_bf16(aq, bk, aD);
    }
#pragma unroll
    for (int r = 0; r < 8; ++r) {
      int t = mt * 16 + r + hi * 8, s = nt * 16 + n;
      float pv = (s <= t) ? aP[r] : 0.f;
      psh[t * C_ + s] = (__bf16)pv;
      float dv = (s <= t) ? aD[r] : 0.f;
#pragma unroll
      for (int off = 1; off < 16; off <<= 1) dv += __shfl_xor(dv, off, 32);
      if (n == 0) atomicAdd(&denom[t], dv);
    }
  }
  __syncthreads();
  // Build K^T over the dead V buffer; finish denominators
  {
    int i = tid >> 2, s0 = (tid & 3) * 16;
#pragma unroll
    for (int j = 0; j < 16; ++j) vshkT[i * 64 + s0 + j] = ksh[(s0 + j) * 64 + i];
  }
  if (tid < C_) {
    float d = denom[tid];
    for (int h = 0; h < H_; ++h) d += (float)qsh[tid * H_ + h] * zsh[h];
    denom[tid] = fmaxf(d, 1e-6f);
  }
  __syncthreads();

  // Phase 2: O = P K + Q Sp^T ; normalize -> osh (bf16)
#pragma unroll
  for (int pi = 0; pi < 2; ++pi) {
    int p = wave + pi * 8, mt = p >> 2, nt = p & 3;
    v8f acc = zero8();
#pragma unroll
    for (int ks = 0; ks < 2; ++ks) {
      v16bf ap = load_a_row(psh + mt * 16 * C_ + ks * 32, C_, lane);
      v16bf bk = load_bT(vshkT + nt * 16 * 64 + ks * 32, 64, lane);  // B = K
      acc = wmma_bf16(ap, bk, acc);
      v16bf aq = load_a_row(qsh + mt * 16 * H_ + ks * 32, H_, lane);
      v16bf bs = load_bT(sps + nt * 16 * H_ + ks * 32, H_, lane);    // B = Sp^T
      acc = wmma_bf16(aq, bs, acc);
    }
#pragma unroll
    for (int r = 0; r < 8; ++r) {
      int t = mt * 16 + r + hi * 8, i = nt * 16 + n;
      osh[t * H_ + i] = (__bf16)(acc[r] / denom[t]);
    }
  }
  __syncthreads();

  // Phase 3: out tile (64 x 1024) = osh @ W_o, 16 column groups of 64
  for (int g = 0; g < D_ / 64; ++g) {
    {  // stage W_o[0:64, g*64 + 0:64] transposed (N-major)
      int h = tid >> 2, n0 = (tid & 3) * 16;
      const uint4* wp = (const uint4*)(wob + (size_t)h * D_ + g * 64 + n0);
      uint4 w0 = wp[0], w1 = wp[1];
      const __bf16* e0 = (const __bf16*)&w0;
      const __bf16* e1 = (const __bf16*)&w1;
#pragma unroll
      for (int j = 0; j < 8; ++j) wshT[(n0 + j) * 64 + h] = e0[j];
#pragma unroll
      for (int j = 0; j < 8; ++j) wshT[(n0 + 8 + j) * 64 + h] = e1[j];
    }
    __syncthreads();
#pragma unroll
    for (int ti = 0; ti < 2; ++ti) {
      int p = wave * 2 + ti, mt = p >> 2, nt = p & 3;
      v8f acc = zero8();
#pragma unroll
      for (int ks = 0; ks < 2; ++ks) {
        v16bf ao = load_a_row(osh + mt * 16 * H_ + ks * 32, H_, lane);
        v16bf bw = load_bT(wshT + nt * 16 * 64 + ks * 32, 64, lane);
        acc = wmma_bf16(ao, bw, acc);
      }
#pragma unroll
      for (int r = 0; r < 8; ++r) {
        int row = blk * 64 + mt * 16 + r + hi * 8;
        int col = g * 64 + nt * 16 + n;
        outp[(size_t)row * D_ + col] = acc[r];
      }
    }
    __syncthreads();
  }
}

// ---------------------------------------------------------------------------
extern "C" void kernel_launch(void* const* d_in, const int* in_sizes, int n_in,
                              void* d_out, int out_size, void* d_ws, size_t ws_size,
                              hipStream_t stream) {
  (void)in_sizes; (void)n_in; (void)out_size; (void)ws_size;
  const float* x  = (const float*)d_in[0];
  const float* Wq = (const float*)d_in[1];
  const float* Wk = (const float*)d_in[2];
  const float* Wv = (const float*)d_in[3];
  const float* Wo = (const float*)d_in[4];
  float* outp = (float*)d_out;

  char* w = (char*)d_ws;
  auto alloc = [&](size_t bytes) -> char* {
    char* p = w;
    w += (bytes + 255) & ~(size_t)255;
    return p;
  };
  __bf16* qb  = (__bf16*)alloc((size_t)ROWS_ * H_ * 2);
  __bf16* kb  = (__bf16*)alloc((size_t)ROWS_ * H_ * 2);
  __bf16* vb  = (__bf16*)alloc((size_t)ROWS_ * H_ * 2);
  __bf16* wqb = (__bf16*)alloc((size_t)D_ * H_ * 2);
  __bf16* wkb = (__bf16*)alloc((size_t)D_ * H_ * 2);
  __bf16* wvb = (__bf16*)alloc((size_t)D_ * H_ * 2);
  __bf16* wob = (__bf16*)alloc((size_t)H_ * D_ * 2);
  float*  Sc  = (float*)alloc((size_t)B_ * NC_ * H_ * H_ * 4);
  float*  zc  = (float*)alloc((size_t)B_ * NC_ * H_ * 4);
  __bf16* Sp  = (__bf16*)alloc((size_t)B_ * NC_ * H_ * H_ * 2);
  float*  zp  = (float*)alloc((size_t)B_ * NC_ * H_ * 4);

  convert_weights_kernel<<<(D_ * H_ + 255) / 256, 256, 0, stream>>>(
      Wq, Wk, Wv, Wo, wqb, wkb, wvb, wob);
  qkv_kernel<<<ROWS_ / 64, 256, 0, stream>>>(x, wqb, wkb, wvb, qb, kb, vb);
  chunk_state_kernel<<<B_ * NC_, 256, 0, stream>>>(kb, vb, Sc, zc);
  prefix_kernel<<<B_, 256, 0, stream>>>(Sc, zc, Sp, zp);
  chunk_out_kernel<<<B_ * NC_, 256, 0, stream>>>(qb, kb, vb, Sp, zp, wob, outp);
}
